// AffinityHead_48816598286782
// MI455X (gfx1250) — matrix operand order repaired and checked by reading, not compiled, not run on hardware
//
#include <hip/hip_runtime.h>
#include <hip/hip_bf16.h>
#include <math.h>

typedef __attribute__((ext_vector_type(16))) _Float16 v16h;
typedef __attribute__((ext_vector_type(8)))  _Float16 v8h;
typedef __attribute__((ext_vector_type(8)))  float    v8f;

#define HID 128
#define NIMG 4
#define NTOK 128
#define NATOM 1024
#define NGRAPH 64

static __device__ __forceinline__ float sigmoidf_(float x) {
    return 1.0f / (1.0f + __expf(-x));
}
static __device__ __forceinline__ float siluf_(float x) {
    return x / (1.0f + __expf(-x));
}
static __device__ __forceinline__ float lreluf_(float x) {
    return x > 0.0f ? x : 0.01f * x;
}

static __device__ __forceinline__ v8f wmma_f16(v16h a, v16h b, v8f c) {
    // v_wmma_f32_16x16x32_f16: D = A(16x32 f16) * B(32x16 f16) + C(16x16 f32)
    return __builtin_amdgcn_wmma_f32_16x16x32_f16(
        /*neg_a=*/false, a, /*neg_b=*/false, b,
        /*c_mod=*/(short)0, c, /*reuse_a=*/false, /*reuse_b=*/false);
}

// Load 16 halves via two 16B chunks (LDS pitch is only 16B aligned).
static __device__ __forceinline__ v16h ld16h(const _Float16* p) {
    v8h lo = *(const v8h*)(p);
    v8h hi = *(const v8h*)(p + 8);
    v16h r;
#pragma unroll
    for (int i = 0; i < 8; ++i) { r[i] = lo[i]; r[i + 8] = hi[i]; }
    return r;
}

// ---------------- workspace layout (byte offsets) ----------------
#define WS_TOK    0u          // 4*128*128 f32            = 262144 B
#define WS_ATOME  262144u     // 4*1024 f32               = 16384 B
#define WS_S0     278528u     // 4*32*27 f32              = 13824 B
#define WS_S1     292352u     // 4*64*27 f32              = 27648 B
#define WS_P0     320000u     // 4*128 f32
#define WS_P1     322048u     // 4*128 f32
#define WS_PF     324096u     // 4*128 f32
#define WS_BIAS   326144u     // 4*64 f32
#define WS_BLOB   327168u     // 4*64*4*32*16 f16         = 1048576 B (32B aligned)

// ---------------- small kernels ----------------

__global__ void k_zero(float* p, int n) {
    int i = blockIdx.x * blockDim.x + threadIdx.x;
    if (i < n) p[i] = 0.0f;
}

// tok[n,t,c] = silu(token_features[n,t,:]) @ W_token + b_token
__global__ void k_tok(const float* __restrict__ tf, const float* __restrict__ Wt,
                      const float* __restrict__ bt, float* __restrict__ tok) {
    int t = blockIdx.x, n = blockIdx.y, c = threadIdx.x;
    __shared__ float xs[256];
    for (int i = threadIdx.x; i < 256; i += blockDim.x)
        xs[i] = siluf_(tf[((size_t)(n * NTOK + t)) * 256 + i]);
    __syncthreads();
    float s = bt[c];
    for (int k = 0; k < 256; ++k) s += xs[k] * Wt[k * HID + c];
    tok[((size_t)(n * NTOK + t)) * HID + c] = s;
}

// atoms = lig_atom @ W_atom + b_atom; pack f16 into WMMA-A layout blob.
// blob[((n*64+tile)*4+kstep)*32+lane][16 halves], lane = M + 16*g,
// h = K32%16, g = K32/16 where c = kstep*32 + K32, M = a%16.
__global__ void k_atoms(const float* __restrict__ la, const float* __restrict__ Wa,
                        const float* __restrict__ ba, _Float16* __restrict__ blob) {
    int n = blockIdx.x >> 6, tile = blockIdx.x & 63;
    __shared__ float xs[16 * 64];
    for (int i = threadIdx.x; i < 1024; i += blockDim.x)
        xs[i] = la[((size_t)(n * NATOM + tile * 16)) * 64 + i];
    __syncthreads();
    for (int idx = threadIdx.x; idx < 2048; idx += blockDim.x) {
        int m = idx >> 7, c = idx & 127;
        float s = ba[c];
        for (int k = 0; k < 64; ++k) s += xs[m * 64 + k] * Wa[k * HID + c];
        int kstep = c >> 5, K32 = c & 31;
        int h = K32 & 15;
        int g = K32 >> 4;
        int lane = m + (g << 4);
        blob[(size_t)((((n * 64 + tile) * 4 + kstep) * 32 + lane)) * 16 + h] = (_Float16)s;
    }
}

// S[n,c,tap] = sum over window positions of silu(X)[n,c, pos+tap]  (3x3x3 taps)
__global__ void k_winsum(const float* __restrict__ X, float* __restrict__ S,
                         int C, int Dim) {
    int c = blockIdx.x, n = blockIdx.y;
    extern __shared__ float xsd[];
    int vol = Dim * Dim * Dim, win = Dim - 2, wv = win * win * win;
    const float* xp = X + ((size_t)(n * C + c)) * vol;
    for (int i = threadIdx.x; i < vol; i += blockDim.x) xsd[i] = siluf_(xp[i]);
    __syncthreads();
    float loc[27];
#pragma unroll
    for (int t = 0; t < 27; ++t) loc[t] = 0.0f;
    for (int i = threadIdx.x; i < wv; i += blockDim.x) {
        int w = i % win, h = (i / win) % win, d = i / (win * win);
        int base = (d * Dim + h) * Dim + w;
#pragma unroll
        for (int kd = 0; kd < 3; ++kd)
#pragma unroll
            for (int kh = 0; kh < 3; ++kh)
#pragma unroll
                for (int kw = 0; kw < 3; ++kw)
                    loc[(kd * 3 + kh) * 3 + kw] += xsd[base + (kd * Dim + kh) * Dim + kw];
    }
    __shared__ float sS[27];
    if (threadIdx.x < 27) sS[threadIdx.x] = 0.0f;
    __syncthreads();
#pragma unroll
    for (int t = 0; t < 27; ++t) atomicAdd(&sS[t], loc[t]);
    __syncthreads();
    if (threadIdx.x < 27) S[((size_t)(n * C + c)) * 27 + threadIdx.x] = sS[threadIdx.x];
}

// p[n,o] = (sum_{c,tap} W[o,c,tap]*S[n,c,tap]) * inv + b[o]
__global__ void k_pconv(const float* __restrict__ S, const float* __restrict__ W,
                        const float* __restrict__ b, float* __restrict__ p,
                        int C, float inv) {
    int idx = blockIdx.x * blockDim.x + threadIdx.x;
    if (idx >= NIMG * HID) return;
    int n = idx >> 7, o = idx & 127;
    const float* Wo = W + (size_t)o * C * 27;
    const float* Sn = S + (size_t)n * C * 27;
    float s = 0.0f;
    for (int i = 0; i < C * 27; ++i) s += Wo[i] * Sn[i];
    p[idx] = s * inv + b[o];
}

// pocket/tok_agg/pf gate. One block per image, 128 threads.
__global__ void k_pf(const float* __restrict__ p0, const float* __restrict__ p1,
                     const float* __restrict__ tok,
                     const float* __restrict__ Wp, const float* __restrict__ bp,
                     const float* __restrict__ Wc, const float* __restrict__ bc,
                     const float* __restrict__ Wg, const float* __restrict__ bg,
                     float* __restrict__ pf) {
    int n = blockIdx.x, c = threadIdx.x;
    __shared__ float pcat[256];
    __shared__ float pfin[384];
    pcat[c] = p0[n * HID + c];
    pcat[128 + c] = p1[n * HID + c];
    float tsum = 0.0f;
    for (int t = 0; t < NTOK; ++t) tsum += tok[((size_t)(n * NTOK + t)) * HID + c];
    __syncthreads();
    float pk = bp[c];
    for (int k = 0; k < 256; ++k) pk += siluf_(pcat[k]) * Wp[k * HID + c];
    pfin[c] = pk;
    pfin[128 + c] = tsum;
    pfin[256 + c] = tsum * (1.0f / NTOK);
    __syncthreads();
    float cat = bc[c], gate = bg[c];
    for (int k = 0; k < 384; ++k) {
        float z = pfin[k];
        cat += z * Wc[k * HID + c];
        gate += z * Wg[k * HID + c];
    }
    pf[n * HID + c] = cat * sigmoidf_(gate);
}

// bias[n,g] = lrelu([pf,gf] @ W_bias1 + b1) @ W_bias2 + b2
__global__ void k_bias(const float* __restrict__ pf, const float* __restrict__ lg,
                       const float* __restrict__ Wg, const float* __restrict__ bg,
                       const float* __restrict__ W1, const float* __restrict__ b1,
                       const float* __restrict__ W2, const float* __restrict__ b2,
                       float* __restrict__ biasw) {
    int n = blockIdx.x >> 6, g = blockIdx.x & 63, c = threadIdx.x;
    __shared__ float zsh[256];
    __shared__ float red[128];
    float gfc = bg[c];
    for (int k = 0; k < 64; ++k)
        gfc += lg[((size_t)(n * NGRAPH + g)) * 64 + k] * Wg[k * HID + c];
    zsh[c] = pf[n * HID + c];
    zsh[128 + c] = gfc;
    __syncthreads();
    float h1 = b1[c];
    for (int k = 0; k < 256; ++k) h1 += zsh[k] * W1[k * HID + c];
    red[c] = lreluf_(h1) * W2[c];
    __syncthreads();
    for (int s = 64; s > 0; s >>= 1) {
        if (c < s) red[c] += red[c + s];
        __syncthreads();
    }
    if (c == 0) biasw[n * NGRAPH + g] = red[0] + b2[0];
}

// ---------------- the big fused WMMA kernel ----------------
// One block per (j, n). Build W_j[k,c] = tok[n,j,k]*W_int[k,c] in LDS (f16),
// then stream 1024 atoms (A blob, contiguous-k A layout) through v_wmma_f32_16x16x32_f16.
__global__ __launch_bounds__(256, 1)
void k_interact(const _Float16* __restrict__ blob, const float* __restrict__ tok,
                const float* __restrict__ W_int, const float* __restrict__ b_int,
                const float* __restrict__ W_pe, const float* __restrict__ b_pe,
                const float* __restrict__ W_pg, const float* __restrict__ b_pg,
                float* __restrict__ atom_e) {
    const int j = blockIdx.x, n = blockIdx.y;
    __shared__ _Float16 Wj[128 * 136];   // [c][k], pitch 136 halves (bank-friendly)
    __shared__ float trow[128], wpe[128], wpg[128], bint[128];

    const int tid = threadIdx.x;
    if (tid < 128) {
        trow[tid] = tok[((size_t)(n * NTOK + j)) * HID + tid];
        wpe[tid] = W_pe[tid];
        wpg[tid] = W_pg[tid];
        bint[tid] = b_int[tid];
    }
    __syncthreads();
    for (int idx = tid; idx < 128 * 128; idx += 256) {
        int k = idx >> 7, c = idx & 127;               // coalesced W_int reads
        Wj[c * 136 + k] = (_Float16)(trow[k] * W_int[k * HID + c]);
    }
    __syncthreads();

    const float bpe = b_pe[0], bpg = b_pg[0];
    const int lane = tid & 31, wave = tid >> 5;
    const int g = lane >> 4, nn = lane & 15;           // D layout: N=nn, M=r+8*g
    const v16h* abase = (const v16h*)blob;

    for (int ti = 0; ti < 8; ++ti) {
        const int tile = ti * 8 + wave;                // 64 tiles over 8 waves
        // A: 4 K-steps, one aligned 32B load per lane each (contiguous-k order, k = 16*g + h)
        v16h a[4];
#pragma unroll
        for (int ks = 0; ks < 4; ++ks)
            a[ks] = abase[(size_t)(((n * 64 + tile) * 4 + ks) * 32 + lane)];

        float spe[8], spg[8];
#pragma unroll
        for (int r = 0; r < 8; ++r) { spe[r] = 0.0f; spg[r] = 0.0f; }

#pragma unroll
        for (int ct = 0; ct < 8; ++ct) {
            const int c = ct * 16 + nn;
            const float bc = bint[c];
            v8f acc = {bc, bc, bc, bc, bc, bc, bc, bc};
            const _Float16* wb = &Wj[c * 136 + g * 16];   // B: lane=N+16*g, h=K%16
            acc = wmma_f16(a[0], ld16h(wb), acc);
            acc = wmma_f16(a[1], ld16h(wb + 32), acc);
            acc = wmma_f16(a[2], ld16h(wb + 64), acc);
            acc = wmma_f16(a[3], ld16h(wb + 96), acc);
            const float wpec = wpe[c], wpgc = wpg[c];
#pragma unroll
            for (int r = 0; r < 8; ++r) {
                float h = lreluf_(acc[r]);
                spe[r] += h * wpec;
                spg[r] += h * wpgc;
            }
        }
        // reduce across the 16 N-lanes of each half-wave
#pragma unroll
        for (int m = 1; m < 16; m <<= 1) {
#pragma unroll
            for (int r = 0; r < 8; ++r) {
                spe[r] += __shfl_xor(spe[r], m, 32);
                spg[r] += __shfl_xor(spg[r], m, 32);
            }
        }
        if (nn == 0) {
#pragma unroll
            for (int r = 0; r < 8; ++r) {
                float pe = (spe[r] + bpe) * sigmoidf_(spg[r] + bpg);
                atomicAdd(&atom_e[(size_t)n * NATOM + tile * 16 + g * 8 + r], pe);
            }
        }
    }
}

// out[n,g] = bias[n,g] + segment_sum(atom_e[n,:], ligand_batch)[g]
__global__ void k_final(const float* __restrict__ atom_e, const int* __restrict__ lb,
                        const float* __restrict__ biasw, float* __restrict__ out) {
    int idx = blockIdx.x * blockDim.x + threadIdx.x;
    if (idx >= NIMG * NGRAPH) return;
    int n = idx >> 6, g = idx & 63;
    float s = 0.0f;
    for (int a = 0; a < NATOM; ++a)
        if (lb[a] == g) s += atom_e[(size_t)n * NATOM + a];
    out[idx] = s + biasw[idx];
}

// ---------------- launcher ----------------
extern "C" void kernel_launch(void* const* d_in, const int* in_sizes, int n_in,
                              void* d_out, int out_size, void* d_ws, size_t ws_size,
                              hipStream_t stream) {
    const float* ms0   = (const float*)d_in[0];
    const float* ms1   = (const float*)d_in[1];
    const float* tf    = (const float*)d_in[2];
    const float* la    = (const float*)d_in[3];
    const float* lg    = (const float*)d_in[4];
    const int*   lb    = (const int*)d_in[5];
    const float* Wtok  = (const float*)d_in[6];
    const float* btok  = (const float*)d_in[7];
    const float* Wc0   = (const float*)d_in[8];
    const float* bc0   = (const float*)d_in[9];
    const float* Wc1   = (const float*)d_in[10];
    const float* bc1   = (const float*)d_in[11];
    const float* Wpock = (const float*)d_in[12];
    const float* bpock = (const float*)d_in[13];
    const float* Wcat  = (const float*)d_in[14];
    const float* bcat  = (const float*)d_in[15];
    const float* Wgate = (const float*)d_in[16];
    const float* bgate = (const float*)d_in[17];
    const float* Watom = (const float*)d_in[18];
    const float* batom = (const float*)d_in[19];
    const float* Wgr   = (const float*)d_in[20];
    const float* bgr   = (const float*)d_in[21];
    const float* Wb1   = (const float*)d_in[22];
    const float* bb1   = (const float*)d_in[23];
    const float* Wb2   = (const float*)d_in[24];
    const float* bb2   = (const float*)d_in[25];
    const float* Wint  = (const float*)d_in[26];
    const float* bint  = (const float*)d_in[27];
    const float* Wpe   = (const float*)d_in[28];
    const float* bpe   = (const float*)d_in[29];
    const float* Wpg   = (const float*)d_in[30];
    const float* bpg   = (const float*)d_in[31];

    char* ws = (char*)d_ws;
    float*    tok    = (float*)(ws + WS_TOK);
    float*    atom_e = (float*)(ws + WS_ATOME);
    float*    S0     = (float*)(ws + WS_S0);
    float*    S1     = (float*)(ws + WS_S1);
    float*    p0     = (float*)(ws + WS_P0);
    float*    p1     = (float*)(ws + WS_P1);
    float*    pf     = (float*)(ws + WS_PF);
    float*    biasw  = (float*)(ws + WS_BIAS);
    _Float16* blob   = (_Float16*)(ws + WS_BLOB);

    // zero per-atom accumulator (harness poisons ws)
    k_zero<<<(NIMG * NATOM + 255) / 256, 256, 0, stream>>>(atom_e, NIMG * NATOM);

    // token projection
    k_tok<<<dim3(NTOK, NIMG), 128, 0, stream>>>(tf, Wtok, btok, tok);

    // atom projection + WMMA-A f16 packing
    k_atoms<<<NIMG * 64, 256, 0, stream>>>(la, Watom, batom, blob);

    // conv+mean collapsed to tapped box-sums then tiny dots
    k_winsum<<<dim3(32, NIMG), 128, 16 * 16 * 16 * sizeof(float), stream>>>(ms0, S0, 32, 16);
    k_winsum<<<dim3(64, NIMG), 128, 8 * 8 * 8 * sizeof(float), stream>>>(ms1, S1, 64, 8);
    k_pconv<<<2, 256, 0, stream>>>(S1, Wc0, bc0, p0, 64, 1.0f / 216.0f);   // p0 <- ms_feat_1
    k_pconv<<<2, 256, 0, stream>>>(S0, Wc1, bc1, p1, 32, 1.0f / 2744.0f);  // p1 <- ms_feat_0

    // pocket + token aggregate + gated fusion
    k_pf<<<NIMG, 128, 0, stream>>>(p0, p1, tok, Wpock, bpock, Wcat, bcat, Wgate, bgate, pf);

    // per-graph bias MLP
    k_bias<<<NIMG * NGRAPH, 128, 0, stream>>>(pf, lg, Wgr, bgr, Wb1, bb1, Wb2, bb2, biasw);

    // the 17-GFLOP fused interaction GEMM (WMMA)
    k_interact<<<dim3(NTOK, NIMG), 256, 0, stream>>>(blob, tok, Wint, bint,
                                                     Wpe, bpe, Wpg, bpg, atom_e);

    // segment-sum + bias -> output [4,64]
    k_final<<<1, 256, 0, stream>>>(atom_e, lb, biasw, (float*)d_out);
}
